// MultiHeadedSelfAttention_22041772163354
// MI455X (gfx1250) — matrix operand order repaired and checked by reading, not compiled, not run on hardware
//
#include <hip/hip_runtime.h>
#include <hip/hip_bf16.h>

// ---------------------------------------------------------------------------
// MHSA for MI455X (gfx1250): bf16 WMMA (16x16x32, f32 accum), wave32,
// async global->LDS tile staging (ASYNCcnt) with double buffering.
//   B=4, T=2048, C=1024, H=16, D=64
// ---------------------------------------------------------------------------

#define B_ 4
#define T_ 2048
#define C_ 1024
#define H_ 16
#define D_ 64

typedef __attribute__((ext_vector_type(16))) __bf16 v16bf;
typedef __attribute__((ext_vector_type(8)))  float  v8f;

union AF {                    // one WMMA bf16 A/B operand (16 x bf16 per lane)
    v16bf v;
    uint4 q[2];
};

__device__ __forceinline__ unsigned short f2bf(float f) {
    unsigned u = __builtin_bit_cast(unsigned, f);
    u += 0x7FFFu + ((u >> 16) & 1u);      // round-to-nearest-even
    return (unsigned short)(u >> 16);
}

__device__ __forceinline__ v8f wmma_bf16(const AF& a, const AF& b, v8f c) {
    return __builtin_amdgcn_wmma_f32_16x16x32_bf16(
        false, a.v, false, b.v, (short)0, c, false, false);
}

// Async copy of 16 bytes global -> LDS per lane (tracked by ASYNCcnt).
__device__ __forceinline__ void async_ld16(unsigned lds_off, const void* g) {
    asm volatile("global_load_async_to_lds_b128 %0, %1, off"
                 :: "v"(lds_off), "v"((unsigned long long)(uintptr_t)g)
                 : "memory");
}
__device__ __forceinline__ void wait_async0() {
    asm volatile("s_wait_asynccnt 0x0" ::: "memory");
}
__device__ __forceinline__ unsigned lds_off(const void* p) {
    return (unsigned)(uintptr_t)p;        // LDS aperture keeps offset in [31:0]
}

// ---------------------------------------------------------------------------
// Kernel 1: f32 -> bf16 (RNE), 4 elements / thread.  Used for x and W.
// ---------------------------------------------------------------------------
__global__ __launch_bounds__(256) void convert_f32_bf16(
    const float* __restrict__ x, unsigned short* __restrict__ y) {
    int i = blockIdx.x * 256 + threadIdx.x;      // float4 index
    float4 f = ((const float4*)x)[i];
    ushort4 o;
    o.x = f2bf(f.x); o.y = f2bf(f.y); o.z = f2bf(f.z); o.w = f2bf(f.w);
    ((ushort4*)y)[i] = o;
}

// ---------------------------------------------------------------------------
// Kernel 2: y = x @ W^T + b.  Block tile 256x128, 8 waves, wave tile 32x128
// (2 A-frags x 8 B-frags = 16 WMMAs per K-step, each B-frag reused twice).
// W tile (bf16) staged via async-to-LDS, double buffered.
// z==0: Q (scaled 1/sqrt(D)) -> [B,H,T,D];  z==1: K -> [B,H,T,D]
// z==2: V -> [B,H,D,T] (transposed for the PV WMMA B-operand)
// ---------------------------------------------------------------------------
__global__ __launch_bounds__(256) void qkv_gemm(
    const unsigned short* __restrict__ xbf,
    const unsigned short* __restrict__ Wqb, const unsigned short* __restrict__ Wkb,
    const unsigned short* __restrict__ Wvb,
    const float* __restrict__ bq, const float* __restrict__ bk,
    const float* __restrict__ bv,
    unsigned short* __restrict__ qh, unsigned short* __restrict__ kh,
    unsigned short* __restrict__ vt) {

    const int z = blockIdx.z;
    const unsigned short* W = (z == 0) ? Wqb : (z == 1) ? Wkb : Wvb;
    const float* bias = (z == 0) ? bq : (z == 1) ? bk : bv;
    unsigned short* outp = (z == 0) ? qh : (z == 1) ? kh : vt;

    const int nbase = blockIdx.x * 128;
    const int mbase = blockIdx.y * 256;
    const int tid  = threadIdx.x;
    const int wave = tid >> 5;
    const int lane = tid & 31;
    const int hl   = lane >> 4;
    const int l16  = lane & 15;

    __shared__ unsigned short Wt[2][128][40];    // double-buffered W tile (+pad)
    const unsigned wt0 = lds_off(&Wt[0][0][0]);
    const int crow = tid >> 2, cseg = tid & 3;   // cooperative copy mapping

    v8f acc[2][8];
#pragma unroll
    for (int g = 0; g < 2; ++g)
#pragma unroll
        for (int j = 0; j < 8; ++j) acc[g][j] = (v8f)(0.0f);

    const int wrow0 = mbase + wave * 32;
    const unsigned short* a0p = xbf + (wrow0 + l16) * C_ + hl * 8;
    const unsigned short* a1p = xbf + (wrow0 + 16 + l16) * C_ + hl * 8;

    // issue async copies of one 128x32 bf16 W tile into buffer `buf`
    auto issueW = [&](int buf, int k0) {
#pragma unroll
        for (int rep = 0; rep < 2; ++rep) {
            const int id  = tid + rep * 256;         // 512 x 16B copies
            const int row = id >> 2, seg = id & 3;
            async_ld16(wt0 + (unsigned)((buf * 128 * 40 + row * 40 + seg * 8) * 2),
                       W + (nbase + row) * C_ + /*k0*/ 0 + seg * 8 + k0);
        }
    };
    (void)crow; (void)cseg;

    issueW(0, 0);
    for (int it = 0; it < C_ / 32; ++it) {
        const int cur = it & 1;
        const int k0  = it * 32;
        wait_async0();
        __syncthreads();
        if (it < C_ / 32 - 1) issueW(cur ^ 1, k0 + 32);

        // A fragments (global, L2-resident bf16 x)
        AF a0, a1;
        a0.q[0] = *(const uint4*)(a0p + k0);
        a0.q[1] = *(const uint4*)(a0p + k0 + 16);
        a1.q[0] = *(const uint4*)(a1p + k0);
        a1.q[1] = *(const uint4*)(a1p + k0 + 16);

        // prefetch-1 pipelined B fragments
        AF bcur, bnxt;
        {
            const unsigned short* bp = &Wt[cur][l16][hl * 8];
            bcur.q[0] = *(const uint4*)bp;
            bcur.q[1] = *(const uint4*)(bp + 16);
        }
#pragma unroll
        for (int j = 0; j < 8; ++j) {
            if (j < 7) {
                const unsigned short* bp = &Wt[cur][(j + 1) * 16 + l16][hl * 8];
                bnxt.q[0] = *(const uint4*)bp;
                bnxt.q[1] = *(const uint4*)(bp + 16);
            }
            acc[0][j] = wmma_bf16(a0, bcur, acc[0][j]);
            acc[1][j] = wmma_bf16(a1, bcur, acc[1][j]);
            bcur = bnxt;
        }
        __syncthreads();   // tile fully consumed before next overwrite
    }

    const float scale = (z == 0) ? 0.125f : 1.0f;   // 1/sqrt(D) folded into Q
#pragma unroll
    for (int j = 0; j < 8; ++j) {
        const int n = nbase + j * 16 + l16;
        const float bb = bias[n];
        const int hh = n >> 6;
        const int dd = n & 63;
#pragma unroll
        for (int g = 0; g < 2; ++g) {
#pragma unroll
            for (int r = 0; r < 8; ++r) {
                const int grow = wrow0 + g * 16 + r + hl * 8;   // [0, B*T)
                const int bi = grow >> 11;
                const int tt = grow & (T_ - 1);
                const unsigned short v16 = f2bf((acc[g][j][r] + bb) * scale);
                if (z < 2)
                    outp[(((bi * H_ + hh) * T_) + tt) * D_ + dd] = v16;
                else
                    outp[(((bi * H_ + hh) * D_) + dd) * T_ + tt] = v16;
            }
        }
    }
}

// ---------------------------------------------------------------------------
// Kernel 3: flash attention.  grid = (T/128, H, B), block = 256 (8 waves).
// Each wave: 16 query rows; 32 keys / iteration; K and V^T tiles staged with
// async-to-LDS, double buffered; online softmax; P relayout via wave-local LDS.
// ---------------------------------------------------------------------------
__global__ __launch_bounds__(256) void attn(
    const unsigned short* __restrict__ qh,
    const unsigned short* __restrict__ kh,
    const unsigned short* __restrict__ vt,
    const int* __restrict__ mask,
    float* __restrict__ out) {

    const int tid  = threadIdx.x;
    const int wave = tid >> 5;
    const int lane = tid & 31;
    const int hl   = lane >> 4;
    const int l16  = lane & 15;
    const int h = blockIdx.y;
    const int b = blockIdx.z;
    const int bh = b * H_ + h;
    const int tq0 = blockIdx.x * 128 + wave * 16;

    __shared__ unsigned short kt[2][32][72];       // 32 keys x 64 d (+pad)
    __shared__ unsigned short vs[2][64][40];       // 64 d x 32 keys (+pad)
    __shared__ unsigned short pbuf[8][16][32];     // per-wave P relayout buffer
    __shared__ int msk[T_];

    const unsigned kt0 = lds_off(&kt[0][0][0]);
    const unsigned vs0 = lds_off(&vs[0][0][0]);

    // stage the whole mask row for this batch once
    for (int i = tid; i < T_ / 4; i += 256) {
        int4 m4 = ((const int4*)(mask + b * T_))[i];
        *(int4*)&msk[i * 4] = m4;
    }

    // Q fragments (A operand), rows tq0+l16, two 32-wide chunks over d
    AF qa[2];
    {
        const unsigned short* qp = qh + (bh * T_ + tq0 + l16) * D_;
#pragma unroll
        for (int c = 0; c < 2; ++c) {
            const unsigned short* p = qp + c * 32 + hl * 8;
            qa[c].q[0] = *(const uint4*)p;
            qa[c].q[1] = *(const uint4*)(p + 16);
        }
    }

    // async tile stage: 256 threads x (1 K-copy + 1 V-copy) x 16B = 8KB
    auto issueKV = [&](int buf, int kb) {
        const int row = tid >> 3, seg = tid & 7;
        async_ld16(kt0 + (unsigned)((buf * 32 * 72 + row * 72 + seg * 8) * 2),
                   kh + (bh * T_ + kb + row) * D_ + seg * 8);
        const int row2 = tid >> 2, seg2 = tid & 3;
        async_ld16(vs0 + (unsigned)((buf * 64 * 40 + row2 * 40 + seg2 * 8) * 2),
                   vt + (bh * D_ + row2) * T_ + kb + seg2 * 8);
    };

    v8f o[4];
#pragma unroll
    for (int j = 0; j < 4; ++j) o[j] = (v8f)(0.0f);
    float mrun[8], lrun[8];
#pragma unroll
    for (int r = 0; r < 8; ++r) { mrun[r] = -3.0e38f; lrun[r] = 0.0f; }

    issueKV(0, 0);
    for (int it = 0; it < T_ / 32; ++it) {
        const int cur = it & 1;
        const int kb  = it * 32;
        wait_async0();
        __syncthreads();
        if (it < T_ / 32 - 1) issueKV(cur ^ 1, kb + 32);

        // S = Q K^T : two 16-key fragments, K-dim = d in two 32-chunks
        v8f s0 = (v8f)(0.0f), s1 = (v8f)(0.0f);
#pragma unroll
        for (int c = 0; c < 2; ++c) {
            AF b0, b1;
            const unsigned short* p0 = &kt[cur][l16][c * 32 + hl * 8];
            b0.q[0] = *(const uint4*)p0; b0.q[1] = *(const uint4*)(p0 + 16);
            const unsigned short* p1 = &kt[cur][16 + l16][c * 32 + hl * 8];
            b1.q[0] = *(const uint4*)p1; b1.q[1] = *(const uint4*)(p1 + 16);
            s0 = wmma_bf16(qa[c], b0, s0);
            s1 = wmma_bf16(qa[c], b1, s1);
        }

        const int m0 = msk[kb + l16];
        const int m1 = msk[kb + 16 + l16];
#pragma unroll
        for (int r = 0; r < 8; ++r) {
            if (!m0) s0[r] = -1.0e30f;
            if (!m1) s1[r] = -1.0e30f;
        }

        // online softmax: row max / correction / exp / row sum
        float mnew[8], corr[8];
#pragma unroll
        for (int r = 0; r < 8; ++r) {
            float v = fmaxf(s0[r], s1[r]);
            for (int off = 1; off < 16; off <<= 1)
                v = fmaxf(v, __shfl_xor(v, off, 32));
            const float mn = fmaxf(mrun[r], v);
            corr[r] = __expf(mrun[r] - mn);
            mnew[r] = mn;
        }
#pragma unroll
        for (int r = 0; r < 8; ++r) {
            const float p0 = __expf(s0[r] - mnew[r]);
            const float p1 = __expf(s1[r] - mnew[r]);
            float rs = p0 + p1;
            for (int off = 1; off < 16; off <<= 1)
                rs += __shfl_xor(rs, off, 32);
            lrun[r] = lrun[r] * corr[r] + rs;
            mrun[r] = mnew[r];
            pbuf[wave][r + hl * 8][l16]      = f2bf(p0);   // C-layout write
            pbuf[wave][r + hl * 8][16 + l16] = f2bf(p1);
        }

#pragma unroll
        for (int j = 0; j < 4; ++j)
#pragma unroll
            for (int r = 0; r < 8; ++r) o[j][r] *= corr[r];

        // reload P in A-fragment layout (wave-local region, DScnt waits only)
        AF pa;
        const unsigned short* pp = &pbuf[wave][l16][hl * 8];
        pa.q[0] = *(const uint4*)pp;
        pa.q[1] = *(const uint4*)(pp + 16);

        // O += P (16x32) x V (32x64): 4 WMMAs over 16-wide d chunks
#pragma unroll
        for (int j = 0; j < 4; ++j) {
            AF vb;
            const unsigned short* vp = &vs[cur][j * 16 + l16][hl * 8];
            vb.q[0] = *(const uint4*)vp;
            vb.q[1] = *(const uint4*)(vp + 16);
            o[j] = wmma_bf16(pa, vb, o[j]);
        }
    }

    // epilogue: normalize and store f32, merge heads
#pragma unroll
    for (int j = 0; j < 4; ++j) {
#pragma unroll
        for (int r = 0; r < 8; ++r) {
            const int t = tq0 + r + hl * 8;
            const float val = o[j][r] / lrun[r];
            out[(b * T_ + t) * C_ + h * 64 + j * 16 + l16] = val;
        }
    }
}

// ---------------------------------------------------------------------------
extern "C" void kernel_launch(void* const* d_in, const int* in_sizes, int n_in,
                              void* d_out, int out_size, void* d_ws, size_t ws_size,
                              hipStream_t stream) {
    const float* x    = (const float*)d_in[0];
    const int*   mask = (const int*)  d_in[1];
    const float* Wq   = (const float*)d_in[2];
    const float* bq   = (const float*)d_in[3];
    const float* Wk   = (const float*)d_in[4];
    const float* bk   = (const float*)d_in[5];
    const float* Wv   = (const float*)d_in[6];
    const float* bv   = (const float*)d_in[7];
    float* out = (float*)d_out;

    const size_t NELEM = (size_t)B_ * T_ * C_;           // 8M
    const size_t WELEM = (size_t)C_ * C_;                // 1M
    unsigned short* xbf = (unsigned short*)d_ws;
    unsigned short* qh  = xbf + NELEM;
    unsigned short* kh  = qh + NELEM;
    unsigned short* vt  = kh + NELEM;
    unsigned short* wqb = vt + NELEM;
    unsigned short* wkb = wqb + WELEM;
    unsigned short* wvb = wkb + WELEM;

    // 1) f32 -> bf16 for x and the three weight matrices
    convert_f32_bf16<<<dim3(NELEM / 1024), dim3(256), 0, stream>>>(x, xbf);
    convert_f32_bf16<<<dim3(WELEM / 1024), dim3(256), 0, stream>>>(Wq, wqb);
    convert_f32_bf16<<<dim3(WELEM / 1024), dim3(256), 0, stream>>>(Wk, wkb);
    convert_f32_bf16<<<dim3(WELEM / 1024), dim3(256), 0, stream>>>(Wv, wvb);

    // 2) QKV projections: grid (N/128, M/256, 3)
    qkv_gemm<<<dim3(C_ / 128, (B_ * T_) / 256, 3), dim3(256), 0, stream>>>(
        xbf, wqb, wkb, wvb, bq, bk, bv, qh, kh, vt);

    // 3) attention: grid (T/128, H, B)
    attn<<<dim3(T_ / 128, H_, B_), dim3(256), 0, stream>>>(qh, kh, vt, mask, out);
}
